// DeformConv2dLite_62414464745680
// MI455X (gfx1250) — compile-verified, matrix operand order checked by reading, not run on hardware
//
#include <hip/hip_runtime.h>
#include <hip/hip_bf16.h>

// ---------------------------------------------------------------------------
// Deformable Conv2d (lite) for MI455X / gfx1250.
// GEMM view: out[o,p] = sum_{K=t*64+c} W[o,K] * S[K,p],  M=64, K=576, N=65536
// WMMA: v_wmma_f32_16x16x32_bf16, fp32 accumulate.
// ---------------------------------------------------------------------------

typedef __attribute__((ext_vector_type(16))) __bf16          v16bf;
typedef __attribute__((ext_vector_type(16))) unsigned short  v16u;
typedef __attribute__((ext_vector_type(8)))  float           v8f;

#define BATCH 4
#define CIN   64
#define COUT  64
#define H     128
#define W     128
#define HW    (H * W)          // 16384
#define TAPS  9
#define KTOT  (TAPS * CIN)     // 576
#define KSTEPS (KTOT / 32)     // 18
#define PIXELS (BATCH * HW)    // 65536

// A/B fragment K index for packed bf16 element j in a lane (CDNA5 ISA,
// 16-bit A 16x32 layout; B uses the mirrored layout with N across lanes).
__device__ __forceinline__ int k_of_elem(int j, int laneHi8) {
    return (j < 8 ? j : j + 8) + laneHi8; // laneHi8 = 8 if lane>=16 else 0
}

__device__ __forceinline__ unsigned short f2bf_rne(float f) {
    unsigned u = __float_as_uint(f);
    unsigned r = u + 0x7FFFu + ((u >> 16) & 1u);
    return (unsigned short)(r >> 16);
}

// ---------------------------------------------------------------------------
// Prep: weight[64][64][3][3] fp32 -> bf16, pre-swizzled into A-fragment order:
//   wA[((ks*4 + mt)*32 + lane)*16 + j],  ks = tap*2 + (c>=32)
// o = mt*16 + lane%16,  c = (ks&1)*32 + k_of_elem(j, lane>=16 ? 8 : 0)
// ---------------------------------------------------------------------------
__global__ void prep_weights_kernel(const float* __restrict__ weight,
                                    unsigned short* __restrict__ wA) {
    int idx = blockIdx.x * blockDim.x + threadIdx.x;
    if (idx >= KSTEPS * 4 * 32 * 16) return;
    int j    = idx & 15;
    int lane = (idx >> 4) & 31;
    int mt   = (idx >> 9) & 3;
    int ks   = idx >> 11;                  // 0..17
    int tap  = ks >> 1;
    int c    = ((ks & 1) << 5) + k_of_elem(j, (lane >= 16) ? 8 : 0);
    int o    = (mt << 4) + (lane & 15);
    float wv = weight[(o * CIN + c) * TAPS + tap];
    wA[idx] = f2bf_rne(wv);
}

// ---------------------------------------------------------------------------
// Main kernel. 128 threads = 4 waves; each wave: 16 pixels x all 64 out-ch.
// LDS stages 3 taps (6 ksteps) of swizzled weights per phase (24 KB).
// ---------------------------------------------------------------------------
__global__ __launch_bounds__(128) void deform_conv_wmma_kernel(
        const float* __restrict__ x,
        const float* __restrict__ offset,
        const unsigned short* __restrict__ wA,
        const float* __restrict__ bias,
        float* __restrict__ out) {

    __shared__ __align__(32) unsigned short sA[6 * 4 * 32 * 16]; // 24576 B

    const int tid  = threadIdx.x;
    const int wave = tid >> 5;
    const int lane = tid & 31;
    const int laneHi8 = (lane >= 16) ? 8 : 0;

    // 16 consecutive pixels per wave -> same (b,h), consecutive w.
    const int p    = (blockIdx.x * 4 + wave) * 16 + (lane & 15);
    const int wpix = p & (W - 1);
    const int hpix = (p >> 7) & (H - 1);
    const int bidx = p >> 14;
    const int hw   = hpix * W + wpix;

    const float* __restrict__ xb   = x + (size_t)bidx * CIN * HW;
    const float* __restrict__ offb = offset + (size_t)bidx * 2 * TAPS * HW;

    v8f acc[4] = {};

    for (int ph = 0; ph < 3; ++ph) {
        __syncthreads();
        // cooperative copy of 6 ksteps of weight fragments (24576 B)
        {
            const uint4* __restrict__ src =
                (const uint4*)(wA + ph * (6 * 4 * 32 * 16));
            uint4* dst = (uint4*)sA;
            #pragma unroll
            for (int i = 0; i < 12; ++i)
                dst[tid + i * 128] = src[tid + i * 128];
        }
        __syncthreads();

        #pragma unroll
        for (int tt = 0; tt < 3; ++tt) {
            const int tap = ph * 3 + tt;

            // Bilinear sampling setup (zero-padding, raw pixel coords).
            const float ox = offb[(2 * tap)     * HW + hw];
            const float oy = offb[(2 * tap + 1) * HW + hw];
            const float pxf = (float)wpix + ox;
            const float pyf = (float)hpix + oy;
            const float x0f = floorf(pxf);
            const float y0f = floorf(pyf);
            const float wx1 = pxf - x0f, wx0 = 1.0f - wx1;
            const float wy1 = pyf - y0f, wy0 = 1.0f - wy1;
            const int ix0 = (int)x0f, iy0 = (int)y0f;
            const int ix1 = ix0 + 1,  iy1 = iy0 + 1;
            const bool vx0 = (unsigned)ix0 < (unsigned)W;
            const bool vx1 = (unsigned)ix1 < (unsigned)W;
            const bool vy0 = (unsigned)iy0 < (unsigned)H;
            const bool vy1 = (unsigned)iy1 < (unsigned)H;
            const float w00 = (vx0 && vy0) ? wx0 * wy0 : 0.0f;
            const float w10 = (vx1 && vy0) ? wx1 * wy0 : 0.0f;
            const float w01 = (vx0 && vy1) ? wx0 * wy1 : 0.0f;
            const float w11 = (vx1 && vy1) ? wx1 * wy1 : 0.0f;
            const int cx0 = min(max(ix0, 0), W - 1);
            const int cx1 = min(max(ix1, 0), W - 1);
            const int cy0 = min(max(iy0, 0), H - 1);
            const int cy1 = min(max(iy1, 0), H - 1);
            const int a00 = cy0 * W + cx0, a10 = cy0 * W + cx1;
            const int a01 = cy1 * W + cx0, a11 = cy1 * W + cx1;

            #pragma unroll
            for (int half = 0; half < 2; ++half) {
                const int c0 = half << 5;
                // Build B fragment (32 K-values x 16 pixels): this lane holds
                // pixel n = lane%16, K per k_of_elem.
                v16u bu;
                #pragma unroll
                for (int j = 0; j < 16; ++j) {
                    const int c = c0 + k_of_elem(j, laneHi8);
                    const float* __restrict__ base = xb + c * HW;
                    float v = base[a00] * w00 + base[a10] * w10
                            + base[a01] * w01 + base[a11] * w11;
                    bu[j] = f2bf_rne(v);
                }
                const v16bf bfrag = __builtin_bit_cast(v16bf, bu);

                const int kslot = tt * 2 + half;
                #pragma unroll
                for (int mt = 0; mt < 4; ++mt) {
                    const v16u au = *(const v16u*)(sA +
                        (((kslot * 4 + mt) * 32 + lane) << 4));
                    const v16bf afrag = __builtin_bit_cast(v16bf, au);
                    acc[mt] = __builtin_amdgcn_wmma_f32_16x16x32_bf16(
                        false, afrag, false, bfrag,
                        (short)0, acc[mt], false, false);
                }
            }
        }
    }

    // Epilogue: C/D layout — lane<16: N=lane, M=r; lane>=16: N=lane-16, M=8+r.
    const int mrow = laneHi8;
    #pragma unroll
    for (int mt = 0; mt < 4; ++mt) {
        #pragma unroll
        for (int r = 0; r < 8; ++r) {
            const int o = mt * 16 + mrow + r;
            out[((size_t)bidx * COUT + o) * HW + hw] = acc[mt][r] + bias[o];
        }
    }
}

// ---------------------------------------------------------------------------
extern "C" void kernel_launch(void* const* d_in, const int* in_sizes, int n_in,
                              void* d_out, int out_size, void* d_ws, size_t ws_size,
                              hipStream_t stream) {
    const float* x      = (const float*)d_in[0];
    const float* offset = (const float*)d_in[1];
    const float* weight = (const float*)d_in[2];
    const float* bias   = (const float*)d_in[3];
    float* out          = (float*)d_out;

    unsigned short* wA = (unsigned short*)d_ws; // 36864 bf16 = 73728 bytes

    // 1) swizzle weights to bf16 A-fragments
    prep_weights_kernel<<<(KSTEPS * 4 * 32 * 16 + 255) / 256, 256, 0, stream>>>(
        weight, wA);

    // 2) main deformable-conv GEMM: 65536 pixels / 64 per block
    deform_conv_wmma_kernel<<<PIXELS / 64, 128, 0, stream>>>(
        x, offset, wA, bias, out);
}
